// ISDALoss_11768210391232
// MI455X (gfx1250) — matrix operand order repaired and checked by reading, not compiled, not run on hardware
//
#include <hip/hip_runtime.h>
#include <hip/hip_bf16.h>

typedef __attribute__((ext_vector_type(16))) _Float16 v16h;
typedef __attribute__((ext_vector_type(8)))  float    v8f;

#define NIMG   8
#define AFEAT  256
#define HH     128
#define WW     128
#define HWSZ   (HH * WW)            // 16384
#define NHW    (NIMG * HWSZ)        // 131072
#define NCLS   19
#define NCLS1  20
#define HT     1024
#define WT     1024

// ws layout (float words)
#define OFF_CNT    0                 // counts[20] (padded to 32)
#define OFF_SUM    32                // sums  [a*20 + c], 5120
#define OFF_SQ     (32 + 5120)       // sumsq [a*20 + c], 5120
#define OFF_COV    (32 + 10240)      // new_cov [c*256 + a], 5120
#define OFF_S      (OFF_COV + 5120)  // S table [l*32 + c], 1024
#define OFF_LAB    (OFF_S + 1024)    // labels int32 [NHW]
#define ACC_WORDS  (32 + 10240)      // region to zero before stats

// ---------------------------------------------------------------------------
// K0: nearest-neighbor label downsample + zero accumulators
// ---------------------------------------------------------------------------
__global__ __launch_bounds__(256) void isda_prep(const long long* __restrict__ target,
                                                 int* __restrict__ labels,
                                                 float* __restrict__ accum) {
    int p = blockIdx.x * 256 + threadIdx.x;
    if (p < ACC_WORDS) accum[p] = 0.0f;
    if (p >= NHW) return;
    int n  = p >> 14;            // / HWSZ
    int hw = p & (HWSZ - 1);
    int h  = hw >> 7;
    int w  = hw & (WW - 1);
    // src_idx = floor(dst * in/out) = dst * 8
    long long t = target[(long long)n * (HT * WT) + (long long)(h * 8) * WT + (w * 8)];
    labels[p] = (int)t;
}

// ---------------------------------------------------------------------------
// K1: per-class sum / sumsq / count reduction over features.
// 128 blocks x 256 threads, 1024 pixels per block (never crosses image
// boundary since 1024 | 16384). Coalesced feature reads, LDS ds_add_f32
// accumulation into [a][class] layout (spreads LDS banks), global float
// atomic flush (10260 entries per block).
// ---------------------------------------------------------------------------
__global__ __launch_bounds__(256) void isda_stats(const float* __restrict__ feats,
                                                  const int* __restrict__ labels,
                                                  float* __restrict__ counts_g,
                                                  float* __restrict__ sums_g,
                                                  float* __restrict__ sumsq_g) {
    __shared__ float s_cnt[NCLS1];
    __shared__ float s_sum[AFEAT * NCLS1];   // 20 KB
    __shared__ float s_sq [AFEAT * NCLS1];   // 20 KB
    const int tid = threadIdx.x;
    for (int i = tid; i < NCLS1; i += 256) s_cnt[i] = 0.0f;
    for (int i = tid; i < AFEAT * NCLS1; i += 256) { s_sum[i] = 0.0f; s_sq[i] = 0.0f; }
    __syncthreads();

    const int p0 = blockIdx.x * 1024;
    const int n  = p0 >> 14;
    const int hwbase = p0 & (HWSZ - 1);

    int lab[4];
#pragma unroll
    for (int j = 0; j < 4; ++j) {
        int l = labels[p0 + tid + j * 256];
        lab[j] = (l == 255) ? 0 : l;
        atomicAdd(&s_cnt[lab[j]], 1.0f);
    }

    for (int a = 0; a < AFEAT; ++a) {
        const float* fptr = feats + (((size_t)(n * AFEAT + a)) << 14) + hwbase;
#pragma unroll
        for (int j = 0; j < 4; ++j) {
            float v = fptr[tid + j * 256];
            int la = a * NCLS1 + lab[j];
            atomicAdd(&s_sum[la], v);
            atomicAdd(&s_sq[la], v * v);
        }
    }
    __syncthreads();

    for (int i = tid; i < NCLS1; i += 256) atomicAdd(&counts_g[i], s_cnt[i]);
    for (int i = tid; i < AFEAT * NCLS1; i += 256) {
        atomicAdd(&sums_g[i],  s_sum[i]);
        atomicAdd(&sumsq_g[i], s_sq[i]);
    }
}

// ---------------------------------------------------------------------------
// K2: finalize per-class mean/var and EMA covariance update.
// grid = 20 classes, block = 256 features.
// var = (sumsq - cnt*ave^2)/cnt_c  ==  sum((f-ave)^2)/cnt_c  exactly (ave = sum/cnt).
// ---------------------------------------------------------------------------
__global__ __launch_bounds__(256) void isda_finalize(const float* __restrict__ counts,
                                                     const float* __restrict__ sums,
                                                     const float* __restrict__ sumsq,
                                                     const float* __restrict__ Ave,
                                                     const float* __restrict__ CoV,
                                                     const float* __restrict__ Amount,
                                                     float* __restrict__ new_cov) {
    int c = blockIdx.x;          // 0..19
    int a = threadIdx.x;         // 0..255
    float cnt  = counts[c];
    float cntc = (cnt == 0.0f) ? 1.0f : cnt;
    float s  = sums [a * NCLS1 + c];
    float sq = sumsq[a * NCLS1 + c];
    float ave = s / cntc;
    float var = (sq - cnt * ave * ave) / cntc;
    float denom = cnt + Amount[c];
    float w = (denom > 0.0f) ? (cnt / denom) : 0.0f;
    float d = Ave[c * AFEAT + a] - ave;
    new_cov[c * AFEAT + a] = CoV[c * AFEAT + a] * (1.0f - w) + var * w
                           + w * (1.0f - w) * d * d;
}

// ---------------------------------------------------------------------------
// K3: S[l,c] = ratio * sum_a CV[l,a]*(W[c,a]-W[l,a])^2 via WMMA.
//   Term1 = CV @ (W^2)^T,  Term2 = (CV .* W_l) @ W^T,  T3[l] = sum CV*W_l^2
//   S = ratio * (Term1 - 2*Term2 + T3)
// One wave (EXEC all ones), M: 2 tiles of 16 (rows 0..31, valid <20),
// N: 2 tiles (cols valid <19), K = 256 = 8 x k32 steps. f16 in, f32 accum.
// ---------------------------------------------------------------------------
__global__ __launch_bounds__(32) void isda_table(const float* __restrict__ cv,
                                                 const float* __restrict__ fcw,
                                                 const int* __restrict__ ratio_p,
                                                 float* __restrict__ S) {
    const int lane = threadIdx.x;
    const int half = lane >> 4;
    const int lm   = lane & 15;
    const float ratio = (float)(*ratio_p);

    __shared__ float t3[32];
    float acc3 = 0.0f;
    if (lane < NCLS) {
        for (int k = 0; k < AFEAT; ++k) {
            float w = fcw[lane * AFEAT + k];
            acc3 += cv[lane * AFEAT + k] * w * w;
        }
    }
    t3[lane] = acc3;
    __syncthreads();

    for (int mt = 0; mt < 2; ++mt) {
        const int row = mt * 16 + lm;                 // A-matrix M = lane&15
        for (int nt = 0; nt < 2; ++nt) {
            const int col = nt * 16 + lm;             // B-matrix N = lane&15
            v8f d1 = {};
            v8f d2 = {};
            for (int kt = 0; kt < 8; ++kt) {
                v16h a1, a2, b1, b2;
#pragma unroll
                for (int e = 0; e < 16; ++e) {
                    // A layout: elem e -> K = ((e>>3)<<4) + half*8 + (e&7)
                    int ka = kt * 32 + ((e >> 3) << 4) + half * 8 + (e & 7);
                    float cvv = (row < NCLS1) ? cv[row * AFEAT + ka] : 0.0f;
                    float wl  = (row < NCLS)  ? fcw[row * AFEAT + ka] : 0.0f;
                    a1[e] = (_Float16)cvv;
                    a2[e] = (_Float16)(cvv * wl);
                    // B layout: elem e -> K = ((e>>3)<<4) + half*8 + (e&7)
                    int kb = kt * 32 + ((e >> 3) << 4) + half * 8 + (e & 7);
                    float wc = (col < NCLS) ? fcw[col * AFEAT + kb] : 0.0f;
                    b1[e] = (_Float16)(wc * wc);
                    b2[e] = (_Float16)wc;
                }
                d1 = __builtin_amdgcn_wmma_f32_16x16x32_f16(
                        false, a1, false, b1, (short)0, d1, false, false);
                d2 = __builtin_amdgcn_wmma_f32_16x16x32_f16(
                        false, a2, false, b2, (short)0, d2, false, false);
            }
#pragma unroll
            for (int r = 0; r < 8; ++r) {
                int orow = mt * 16 + half * 8 + r;    // D: M = r + 8*(lane>>4)
                int ocol = nt * 16 + lm;
                if (orow < NCLS1 && ocol < NCLS) {
                    S[orow * 32 + ocol] = ratio * (d1[r] - 2.0f * d2[r] + t3[orow]);
                }
            }
        }
    }
}

// ---------------------------------------------------------------------------
// K4: out[n,c,h,w] = y[n,c,h,w] + 0.5 * S[l,c] * keep   (streaming, coalesced)
// ---------------------------------------------------------------------------
__global__ __launch_bounds__(256) void isda_augment(const float* __restrict__ y,
                                                    const int* __restrict__ labels,
                                                    const float* __restrict__ S,
                                                    float* __restrict__ out,
                                                    int total) {
    int idx = blockIdx.x * 256 + threadIdx.x;
    if (idx >= total) return;
    int hw = idx & (HWSZ - 1);
    int t  = idx >> 14;                  // n*NCLS + c
    int c  = t % NCLS;
    int n  = t / NCLS;
    int lraw = labels[n * HWSZ + hw];
    float keep = (lraw == 255) ? 0.0f : 1.0f;
    int l = (lraw == 255) ? 0 : lraw;
    out[idx] = y[idx] + 0.5f * S[l * 32 + c] * keep;
}

// ---------------------------------------------------------------------------
extern "C" void kernel_launch(void* const* d_in, const int* in_sizes, int n_in,
                              void* d_out, int out_size, void* d_ws, size_t ws_size,
                              hipStream_t stream) {
    (void)in_sizes; (void)n_in; (void)out_size; (void)ws_size;

    const float*     feats  = (const float*)d_in[0];      // [N,A,H,W]
    const float*     y      = (const float*)d_in[1];      // [N,C,H,W]
    const float*     fcw    = (const float*)d_in[2];      // [C,A]
    const float*     Ave    = (const float*)d_in[3];      // [C+1,A]
    const float*     CoV    = (const float*)d_in[4];      // [C+1,A]
    const float*     Amount = (const float*)d_in[5];      // [C+1]
    const long long* target = (const long long*)d_in[6];  // [N,1024,1024] i64
    const int*       ratio  = (const int*)d_in[7];        // scalar

    float* ws      = (float*)d_ws;
    float* counts  = ws + OFF_CNT;
    float* sums    = ws + OFF_SUM;
    float* sumsq   = ws + OFF_SQ;
    float* new_cov = ws + OFF_COV;
    float* Stab    = ws + OFF_S;
    int*   labels  = (int*)(ws + OFF_LAB);

    isda_prep<<<NHW / 256, 256, 0, stream>>>(target, labels, ws);
    isda_stats<<<NHW / 1024, 256, 0, stream>>>(feats, labels, counts, sums, sumsq);
    isda_finalize<<<NCLS1, 256, 0, stream>>>(counts, sums, sumsq, Ave, CoV, Amount, new_cov);
    isda_table<<<1, 32, 0, stream>>>(new_cov, fcw, ratio, Stab);

    int total = NIMG * NCLS * HWSZ;
    isda_augment<<<(total + 255) / 256, 256, 0, stream>>>(y, labels, Stab,
                                                          (float*)d_out, total);
}
